// Learner_13082470383917
// MI455X (gfx1250) — matrix operand order repaired and checked by reading, not compiled, not run on hardware
//
#include <hip/hip_runtime.h>
#include <hip/hip_bf16.h>

// ---------------------------------------------------------------------------
// CDNA5 (gfx1250) implementation. wave32. Matrix work via
// v_wmma_f32_16x16x32_bf16; A-tile staging via global_load_async_to_lds_b128
// (ASYNCcnt) when the toolchain exposes it.
// ---------------------------------------------------------------------------

typedef __bf16 bf16;
typedef __attribute__((ext_vector_type(16))) __bf16 v16bf;
typedef __attribute__((ext_vector_type(8)))  __bf16 v8bf;
typedef __attribute__((ext_vector_type(8)))  float  v8f;
typedef int v4i __attribute__((__vector_size__(4 * sizeof(int))));

#define NUM_ENTITY 4096
#define NUM_OP     12
#define NUM_STEP   4
#define RANK       3
#define NUM_QUERY  24
#define EMBED      128
#define HIDDEN     128
#define BATCH      64
#define RB         (RANK * BATCH)   // 192 fused GEMM rows
#define WN         128              // n-columns per workgroup in GEMM
#define KC         64               // K-chunk staged in LDS
#define NCHUNK     (NUM_ENTITY / KC)
#define LSTRIDE    72               // padded LDS row stride (bf16), 16B-aligned, bank-rotating

#if defined(__gfx1250__) && __has_builtin(__builtin_amdgcn_global_load_async_to_lds_b128) && \
    __has_builtin(__builtin_amdgcn_s_wait_asynccnt)
#define HAVE_ASYNC 1
#else
#define HAVE_ASYNC 0
#endif

__device__ __forceinline__ bf16  f2bf(float x) { return (bf16)x; }   // native v_cvt, RNE
__device__ __forceinline__ float bf2f(bf16 b)  { return (float)b; }
__device__ __forceinline__ float sigmoidf(float x) { return 1.0f / (1.0f + expf(-x)); }

__device__ __forceinline__ void async_cp16(const void* gsrc, void* lds) {
#if HAVE_ASYNC
    __builtin_amdgcn_global_load_async_to_lds_b128(
        (__attribute__((address_space(1))) v4i*)gsrc,
        (__attribute__((address_space(3))) v4i*)lds, 0, 0);
#else
    *(v8bf*)lds = *(const v8bf*)gsrc;
#endif
}

// ---------------------------------------------------------------------------
// Prep kernels (tiny)
// ---------------------------------------------------------------------------

// xt[t][b][k] = query_emb[(t<3 ? qq[b] : NUM_QUERY)][k]
__global__ void k_prep_xt(const int* __restrict__ qq, const float* __restrict__ qemb,
                          float* __restrict__ xt) {
    int idx = blockIdx.x * blockDim.x + threadIdx.x;
    if (idx >= NUM_STEP * BATCH * EMBED) return;
    int k = idx % EMBED;
    int b = (idx / EMBED) % BATCH;
    int t = idx / (EMBED * BATCH);
    int q = (t < NUM_STEP - 1) ? qq[b] : NUM_QUERY;
    xt[idx] = qemb[q * EMBED + k];
}

// wT[rd][k][gate_row] = bf16(w[rd][gate_row][k])  for ih and hh
__global__ void k_prep_w(const float* __restrict__ w_ih, const float* __restrict__ w_hh,
                         bf16* __restrict__ wTih, bf16* __restrict__ wThh) {
    int idx = blockIdx.x * blockDim.x + threadIdx.x;
    const int per = RANK * 2 * 4 * HIDDEN * EMBED;   // 393216
    if (idx >= 2 * per) return;
    int mat = idx / per;
    int rem = idx % per;
    int rd  = rem / (4 * HIDDEN * EMBED);
    int r2  = rem % (4 * HIDDEN * EMBED);
    int row = r2 / EMBED;          // 0..511
    int k   = r2 % EMBED;          // 0..127
    const float* src = mat ? w_hh : w_ih;
    bf16*        dst = mat ? wThh : wTih;
    dst[(rd * EMBED + k) * (4 * HIDDEN) + row] = f2bf(src[(rd * 4 * HIDDEN + row) * EMBED + k]);
}

// MEM[rank][b][n] = one_hot(tt[b]) (same for every rank at t=0); f32 + bf16
__global__ void k_meminit(const int* __restrict__ tt, float* __restrict__ MEM,
                          bf16* __restrict__ MEMbf) {
    int idx = blockIdx.x * blockDim.x + threadIdx.x;
    if (idx >= RB * NUM_ENTITY) return;
    int n  = idx % NUM_ENTITY;
    int b  = (idx / NUM_ENTITY) % BATCH;
    float v = (n == tt[b]) ? 1.0f : 0.0f;
    MEM[idx]   = v;
    MEMbf[idx] = f2bf(v);
}

// ---------------------------------------------------------------------------
// BiLSTM: one block per (rank, dir, batch-block-of-8). 128 threads (= hidden).
// Batches are recurrence-independent -> no cross-WG sync needed.
// ---------------------------------------------------------------------------
__global__ __launch_bounds__(128) void k_lstm(const float* __restrict__ xt,
                                              const bf16* __restrict__ wTih,
                                              const bf16* __restrict__ wThh,
                                              const float* __restrict__ b_ih,
                                              const float* __restrict__ b_hh,
                                              float* __restrict__ h_out) {
    const int bx   = blockIdx.x;
    const int rank = bx >> 4;
    const int dir  = (bx >> 3) & 1;
    const int bb   = bx & 7;            // batch block (8 batches)
    const int h    = threadIdx.x;       // hidden unit
    const int rd   = rank * 2 + dir;

    __shared__ float hs[8][HIDDEN];
    __shared__ float xs[8][EMBED];

    for (int i = h; i < 8 * HIDDEN; i += 128) hs[i >> 7][i & 127] = 0.0f;

    float c[8];
#pragma unroll
    for (int i = 0; i < 8; ++i) c[i] = 0.0f;

    float bias[4];
#pragma unroll
    for (int g = 0; g < 4; ++g)
        bias[g] = b_ih[rd * 4 * HIDDEN + g * HIDDEN + h] + b_hh[rd * 4 * HIDDEN + g * HIDDEN + h];

    const bf16* wi = &wTih[rd * EMBED * 4 * HIDDEN];
    const bf16* wh = &wThh[rd * EMBED * 4 * HIDDEN];
    __syncthreads();

    for (int step = 0; step < NUM_STEP; ++step) {
        const int t = dir ? (NUM_STEP - 1 - step) : step;
        for (int i = h; i < 8 * EMBED; i += 128) {
            int bl = i >> 7, k = i & 127;
            xs[bl][k] = xt[(t * BATCH + (bb * 8 + bl)) * EMBED + k];
        }
        __syncthreads();

        float acc[8][4];
#pragma unroll
        for (int bl = 0; bl < 8; ++bl)
#pragma unroll
            for (int g = 0; g < 4; ++g) acc[bl][g] = bias[g];

        for (int k = 0; k < EMBED; ++k) {
            float wxv[4], whv[4];
#pragma unroll
            for (int g = 0; g < 4; ++g) {
                wxv[g] = bf2f(wi[k * 4 * HIDDEN + g * HIDDEN + h]);
                whv[g] = bf2f(wh[k * 4 * HIDDEN + g * HIDDEN + h]);
            }
#pragma unroll
            for (int bl = 0; bl < 8; ++bl) {
                float xv = xs[bl][k], hv = hs[bl][k];
#pragma unroll
                for (int g = 0; g < 4; ++g) acc[bl][g] += xv * wxv[g] + hv * whv[g];
            }
        }
        __syncthreads();   // everyone done reading hs
#pragma unroll
        for (int bl = 0; bl < 8; ++bl) {
            float ig = sigmoidf(acc[bl][0]);
            float fg = sigmoidf(acc[bl][1]);
            float gg = tanhf(acc[bl][2]);
            float og = sigmoidf(acc[bl][3]);
            c[bl] = fg * c[bl] + ig * gg;
            float hn = og * tanhf(c[bl]);
            hs[bl][h] = hn;
            h_out[((rank * NUM_STEP + t) * BATCH + bb * 8 + bl) * (2 * HIDDEN) + dir * HIDDEN + h] = hn;
        }
        __syncthreads();
    }
}

// ---------------------------------------------------------------------------
// Attention: softmax(h @ lin_w^T + lin_b); one thread per (rank,t,b).
// ---------------------------------------------------------------------------
__global__ void k_attn(const float* __restrict__ h_out, const float* __restrict__ lin_w,
                       const float* __restrict__ lin_b, float* __restrict__ attn_out) {
    int idx = blockIdx.x * blockDim.x + threadIdx.x;
    if (idx >= RANK * NUM_STEP * BATCH) return;
    const float* hp = &h_out[idx * (2 * HIDDEN)];
    float lg[NUM_OP + 1];
    float mx = -1e30f;
#pragma unroll
    for (int o = 0; o < NUM_OP + 1; ++o) {
        float s = lin_b[o];
        for (int k = 0; k < 2 * HIDDEN; ++k) s += hp[k] * lin_w[o * (2 * HIDDEN) + k];
        lg[o] = s;
        mx = fmaxf(mx, s);
    }
    float sum = 0.0f;
#pragma unroll
    for (int o = 0; o < NUM_OP + 1; ++o) { lg[o] = expf(lg[o] - mx); sum += lg[o]; }
    float inv = 1.0f / sum;
#pragma unroll
    for (int o = 0; o < NUM_OP + 1; ++o) attn_out[idx * (NUM_OP + 1) + o] = lg[o] * inv;
}

// ---------------------------------------------------------------------------
// Fused-rank GEMM pass:  PROD[rb][r][n] = sum_m mdb[r][n][m] * MEM[rb][m]
//   One WG per (r_op, 128-wide n-chunk): 12 * 32 = 384 WGs, 256 threads (8 waves).
//   Pipelined: B-tile (mdb, f32->bf16) global loads for chunk kc+1 are issued
//   before the WMMA block of chunk kc; A-tile (bf16 copy) staged async-to-LDS.
// ---------------------------------------------------------------------------
__global__ __launch_bounds__(256) void k_gemm(const float* __restrict__ mdb,
                                              const bf16* __restrict__ Abf,
                                              float* __restrict__ prod) {
    __shared__ bf16 Alds[RB][LSTRIDE];   // 192*72*2 = 27648 B
    __shared__ bf16 Blds[WN][LSTRIDE];   // 128*72*2 = 18432 B

    const int nchunksN = NUM_ENTITY / WN;             // 32
    const int rop = blockIdx.x / nchunksN;            // 0..11
    const int n0  = (blockIdx.x % nchunksN) * WN;
    const int tid = threadIdx.x;
    const int lane  = tid & 31;
    const int wv    = tid >> 5;                       // wave id: owns n-tile wv
    const int lhalf = lane >> 4;
    const int l16   = lane & 15;
    const long mdb_base = (long)rop * NUM_ENTITY * NUM_ENTITY;

    v8f acc[12] = {};
    float4 breg[8];                                   // in-flight B tile (next chunk)

    // ---- prologue: issue chunk 0 ----
#pragma unroll
    for (int j = 0; j < 8; ++j) {                     // B: 128 rows x 64 cols f32
        int i = tid + j * 256;
        int row = i >> 4, c4 = i & 15;
        breg[j] = *(const float4*)&mdb[mdb_base + (long)(n0 + row) * NUM_ENTITY + c4 * 4];
    }
#pragma unroll
    for (int j = 0; j < 6; ++j) {                     // A: 192 rows x 64 cols bf16
        int i = tid + j * 256;
        int row = i >> 3, c8 = i & 7;
        async_cp16(&Abf[row * NUM_ENTITY + c8 * 8], &Alds[row][c8 * 8]);
    }

    for (int kc = 0; kc < NCHUNK; ++kc) {
        // ---- convert + store B(kc) to LDS (waits the in-flight global loads) ----
#pragma unroll
        for (int j = 0; j < 8; ++j) {
            int i = tid + j * 256;
            int row = i >> 4, c4 = i & 15;
            bf16* d = &Blds[row][c4 * 4];
            d[0] = f2bf(breg[j].x); d[1] = f2bf(breg[j].y);
            d[2] = f2bf(breg[j].z); d[3] = f2bf(breg[j].w);
        }
#if HAVE_ASYNC
        __builtin_amdgcn_s_wait_asynccnt(0);          // A(kc) landed in LDS
#endif
        __syncthreads();

        // ---- issue B(kc+1) global loads; latency hides behind the WMMAs ----
        if (kc + 1 < NCHUNK) {
#pragma unroll
            for (int j = 0; j < 8; ++j) {
                int i = tid + j * 256;
                int row = i >> 4, c4 = i & 15;
                breg[j] = *(const float4*)&mdb[mdb_base + (long)(n0 + row) * NUM_ENTITY +
                                               (kc + 1) * KC + c4 * 4];
            }
        }

        // ---- compute: 2 wmma k-steps, A-frags double-buffered in registers ----
#pragma unroll
        for (int ks = 0; ks < KC / 32; ++ks) {
            const int kb = ks * 32;
            v16bf bfrag;
            {   // B frag (32x16): lane l16 = N, K = kb + lhalf*16 + [0..15]
                const v8bf b0 = *(const v8bf*)&Blds[wv * 16 + l16][kb + lhalf * 16];
                const v8bf b1 = *(const v8bf*)&Blds[wv * 16 + l16][kb + lhalf * 16 + 8];
#pragma unroll
                for (int q = 0; q < 8; ++q) { bfrag[q] = b0[q]; bfrag[8 + q] = b1[q]; }
            }
            v16bf afA;
            {   // A frag (16x32): lane l16 = M, K = kb + lhalf*16 + [0..15]
                const v8bf a0 = *(const v8bf*)&Alds[l16][kb + lhalf * 16];
                const v8bf a1 = *(const v8bf*)&Alds[l16][kb + lhalf * 16 + 8];
#pragma unroll
                for (int q = 0; q < 8; ++q) { afA[q] = a0[q]; afA[8 + q] = a1[q]; }
            }
#pragma unroll
            for (int mt = 0; mt < 12; ++mt) {
                v16bf afN;
                if (mt < 11) {                         // prefetch next M-tile's A frag
                    const v8bf a0 = *(const v8bf*)&Alds[(mt + 1) * 16 + l16][kb + lhalf * 16];
                    const v8bf a1 = *(const v8bf*)&Alds[(mt + 1) * 16 + l16][kb + lhalf * 16 + 8];
#pragma unroll
                    for (int q = 0; q < 8; ++q) { afN[q] = a0[q]; afN[8 + q] = a1[q]; }
                }
                acc[mt] = __builtin_amdgcn_wmma_f32_16x16x32_bf16(
                    false, afA, false, bfrag, (short)0, acc[mt], false, false);
                afA = afN;
            }
        }
        __syncthreads();                               // all waves done reading LDS

        // ---- issue A(kc+1) async copy into LDS (safe: reads drained) ----
        if (kc + 1 < NCHUNK) {
#pragma unroll
            for (int j = 0; j < 6; ++j) {
                int i = tid + j * 256;
                int row = i >> 3, c8 = i & 7;
                async_cp16(&Abf[row * NUM_ENTITY + (kc + 1) * KC + c8 * 8], &Alds[row][c8 * 8]);
            }
        }
    }

    // ---- epilogue: C/D layout lane = N, VGPR j -> M = j + lhalf*8 ----
    const int ncol = n0 + wv * 16 + l16;
#pragma unroll
    for (int mt = 0; mt < 12; ++mt)
#pragma unroll
        for (int j = 0; j < 8; ++j) {
            int rb = mt * 16 + lhalf * 8 + j;
            prod[((long)rb * NUM_OP + rop) * NUM_ENTITY + ncol] = acc[mt][j];
        }
}

// ---------------------------------------------------------------------------
// Combine + L1 normalize. One block per (rank, b) = 192 blocks, 256 threads.
// ---------------------------------------------------------------------------
__global__ __launch_bounds__(256) void k_combine(const float* __restrict__ prod,
                                                 const float* __restrict__ attn_all,
                                                 int t,
                                                 float* __restrict__ MEM,
                                                 bf16* __restrict__ MEMbf) {
    const int rb   = blockIdx.x;
    const int rank = rb / BATCH;
    const int b    = rb % BATCH;
    const int tid  = threadIdx.x;

    __shared__ float a_s[NUM_OP + 1];
    __shared__ float red[256];
    if (tid < NUM_OP + 1)
        a_s[tid] = attn_all[((rank * NUM_STEP + t) * BATCH + b) * (NUM_OP + 1) + tid];
    __syncthreads();

    float vals[NUM_ENTITY / 256];
    float s = 0.0f;
#pragma unroll
    for (int i = 0; i < NUM_ENTITY / 256; ++i) {
        int n = tid + i * 256;
        float cmb = MEM[(long)rb * NUM_ENTITY + n] * a_s[NUM_OP];
#pragma unroll
        for (int r = 0; r < NUM_OP; ++r)
            cmb += prod[((long)rb * NUM_OP + r) * NUM_ENTITY + n] * a_s[r];
        vals[i] = cmb;
        s += fabsf(cmb);
    }
    red[tid] = s;
    __syncthreads();
    for (int ofs = 128; ofs > 0; ofs >>= 1) {
        if (tid < ofs) red[tid] += red[tid + ofs];
        __syncthreads();
    }
    const float inv = 1.0f / fmaxf(red[0], 1e-20f);
#pragma unroll
    for (int i = 0; i < NUM_ENTITY / 256; ++i) {
        int n = tid + i * 256;
        float m = vals[i] * inv;
        MEM[(long)rb * NUM_ENTITY + n]   = m;
        MEMbf[(long)rb * NUM_ENTITY + n] = f2bf(m);
    }
}

// prediction[b][n] = sum_rank MEM[rank][b][n]
__global__ void k_final(const float* __restrict__ MEM, float* __restrict__ out) {
    int idx = blockIdx.x * blockDim.x + threadIdx.x;
    if (idx >= BATCH * NUM_ENTITY) return;
    int n = idx % NUM_ENTITY;
    int b = idx / NUM_ENTITY;
    float s = 0.0f;
#pragma unroll
    for (int r = 0; r < RANK; ++r)
        s += MEM[((long)(r * BATCH + b)) * NUM_ENTITY + n];
    out[idx] = s;
}

// ---------------------------------------------------------------------------
// Host launcher
// ---------------------------------------------------------------------------
extern "C" void kernel_launch(void* const* d_in, const int* in_sizes, int n_in,
                              void* d_out, int out_size, void* d_ws, size_t ws_size,
                              hipStream_t stream) {
    const int*   qq    = (const int*)  d_in[0];
    const int*   tt    = (const int*)  d_in[1];
    const float* mdb   = (const float*)d_in[2];
    const float* qemb  = (const float*)d_in[3];
    const float* w_ih  = (const float*)d_in[4];
    const float* w_hh  = (const float*)d_in[5];
    const float* b_ih  = (const float*)d_in[6];
    const float* b_hh  = (const float*)d_in[7];
    const float* lin_w = (const float*)d_in[8];
    const float* lin_b = (const float*)d_in[9];

    char* ws = (char*)d_ws;
    // workspace layout (bytes)
    float* xt    = (float*)(ws + 0);                       // 131072
    bf16*  wTih  = (bf16*) (ws + 131072);                  // 786432
    bf16*  wThh  = (bf16*) (ws + 917504);                  // 786432
    float* hout  = (float*)(ws + 1703936);                 // 786432
    float* MEM   = (float*)(ws + 2490368);                 // 3145728
    bf16*  MEMbf = (bf16*) (ws + 5636096);                 // 1572864
    float* PROD  = (float*)(ws + 7208960);                 // 37748736  (~45 MB total)

    float* out_pred = (float*)d_out;
    float* out_attn = out_pred + BATCH * NUM_ENTITY;       // [3][4][64][13]

    k_prep_xt<<<(NUM_STEP * BATCH * EMBED + 255) / 256, 256, 0, stream>>>(qq, qemb, xt);
    k_prep_w <<<(2 * RANK * 2 * 4 * HIDDEN * EMBED + 255) / 256, 256, 0, stream>>>(w_ih, w_hh, wTih, wThh);
    k_meminit<<<(RB * NUM_ENTITY + 255) / 256, 256, 0, stream>>>(tt, MEM, MEMbf);
    k_lstm   <<<RANK * 2 * 8, 128, 0, stream>>>(xt, wTih, wThh, b_ih, b_hh, hout);
    k_attn   <<<(RANK * NUM_STEP * BATCH + 255) / 256, 256, 0, stream>>>(hout, lin_w, lin_b, out_attn);

    const int gemm_grid = NUM_OP * (NUM_ENTITY / WN);      // 384
    for (int t = 0; t < NUM_STEP - 1; ++t) {
        k_gemm   <<<gemm_grid, 256, 0, stream>>>(mdb, MEMbf, PROD);
        k_combine<<<RB, 256, 0, stream>>>(PROD, out_attn, t, MEM, MEMbf);
    }
    k_final<<<(BATCH * NUM_ENTITY + 255) / 256, 256, 0, stream>>>(MEM, out_pred);
}
